// BertCrf_7610682048908
// MI455X (gfx1250) — matrix operand order repaired and checked by reading, not compiled, not run on hardware
//
#include <hip/hip_runtime.h>

#define Bc 16
#define Lc 256
#define Hc 768
#define NLc 4
#define NHc 12
#define Fc 3072
#define Tc 33
#define TPc 48
#define Mrows (Bc * Lc)
#define HDc 64

typedef __attribute__((ext_vector_type(16))) __bf16 v16bf;
typedef __attribute__((ext_vector_type(8)))  __bf16 v8bf;
typedef __attribute__((ext_vector_type(8)))  float  v8f;

// ---------------------------------------------------------------- WMMA helpers

__device__ __forceinline__ v8f wmma_bf16(v16bf a, v16bf b, v8f c) {
  // D(f32 16x16) = A(bf16 16x32) * B(bf16 32x16) + C
  return __builtin_amdgcn_wmma_f32_16x16x32_bf16(false, a, false, b, (short)0, c,
                                                 false, false);
}

// A-fragment (16x32 bf16 row-major): lane<16 row=lane, K={kb..kb+15},
// kb = kstep+0 ; lane>=16 row=lane-16, kb = kstep+16 (contiguous, matches B)
__device__ __forceinline__ v16bf load_afrag(const __bf16* rowp, int kb) {
  v8bf lo = *(const v8bf*)(rowp + kb);
  v8bf hi = *(const v8bf*)(rowp + kb + 8);
  v16bf f;
#pragma unroll
  for (int j = 0; j < 8; ++j) { f[j] = lo[j]; f[8 + j] = hi[j]; }
  return f;
}

// ---------------------------------------------------------------- CDNA5 async copy
// global -> LDS direct copy, tracked by ASYNCcnt (no VGPR round trip).
__device__ __forceinline__ void async_load_b128(unsigned lds_off, const void* gptr) {
  asm volatile("global_load_async_to_lds_b128 %0, %1, off"
               :
               : "v"(lds_off), "v"((unsigned long long)(uintptr_t)gptr)
               : "memory");
}
__device__ __forceinline__ void wait_asynccnt0() {
  asm volatile("s_wait_asynccnt 0" ::: "memory");
}

// ---------------------------------------------------------------- weight prep

// Wt[n*K + k] = (bf16) W[k*N + n]   (transpose so B-fragments are contiguous)
__global__ void transpose_cast(const float* __restrict__ W, __bf16* __restrict__ Wt,
                               int K, int N) {
  int n = blockIdx.x;
  for (int k = threadIdx.x; k < K; k += blockDim.x)
    Wt[(size_t)n * K + k] = (__bf16)W[(size_t)k * N + n];
}

// classifier: pad N from 33 to 48, transpose, pad bias
__global__ void cls_prep(const float* __restrict__ W, const float* __restrict__ bsrc,
                         __bf16* __restrict__ Wt, float* __restrict__ bpad) {
  int n = blockIdx.x;  // 0..47
  for (int k = threadIdx.x; k < Hc; k += blockDim.x)
    Wt[(size_t)n * Hc + k] = (__bf16)((n < Tc) ? W[(size_t)k * Tc + n] : 0.f);
  if (blockIdx.x == 0 && threadIdx.x < TPc)
    bpad[threadIdx.x] = (threadIdx.x < Tc) ? bsrc[threadIdx.x] : 0.f;
}

// ---------------------------------------------------------------- generic WMMA GEMM
// C[M, Ntiles*16] = A[M,K](bf16) * Wt[N][K](bf16)^T + bias.
// One wave computes a 16 x (16*NT) strip: A-fragment loaded once per k-step and
// reused across NT chained WMMAs (4x better fragment reuse than 1 tile/wave).
// MODE: 0 = store f32, 1 = store bf16, 2 = exact GELU then store bf16
template <int MODE, int NT>
__global__ void gemm_wmma(const __bf16* __restrict__ A, const __bf16* __restrict__ Wt,
                          const float* __restrict__ bias, float* __restrict__ Cf,
                          __bf16* __restrict__ Cb, int K, int Ntiles) {
  const int lane = threadIdx.x & 31;
  const int wave = threadIdx.x >> 5;
  const int g = blockIdx.x * (blockDim.x >> 5) + wave;
  const int ngroups = Ntiles / NT;
  const int tn0 = (g % ngroups) * NT;
  const int tm = g / ngroups;
  const int r15 = lane & 15;
  const int hsel = (lane >> 4) & 1;
  const __bf16* arow = A + (size_t)(tm * 16 + r15) * K;
  const __bf16* brow0 = Wt + (size_t)(tn0 * 16 + r15) * K;
  v8f acc[NT];
#pragma unroll
  for (int t = 0; t < NT; ++t) acc[t] = (v8f){};
  for (int k0 = 0; k0 < K; k0 += 32) {
    v16bf af = load_afrag(arow, k0 + hsel * 16);
#pragma unroll
    for (int t = 0; t < NT; ++t) {
      v16bf bf = *(const v16bf*)(brow0 + (size_t)t * 16 * K + k0 + hsel * 16);
      acc[t] = wmma_bf16(af, bf, acc[t]);
    }
  }
  const int ldc = Ntiles * 16;
  const int mb = tm * 16 + hsel * 8;
#pragma unroll
  for (int t = 0; t < NT; ++t) {
    const int n = (tn0 + t) * 16 + r15;
    const float bs = bias ? bias[n] : 0.f;
#pragma unroll
    for (int r = 0; r < 8; ++r) {
      float v = acc[t][r] + bs;
      if (MODE == 2) v = 0.5f * v * (1.f + erff(v * 0.70710678118654752f));
      size_t idx = (size_t)(mb + r) * ldc + n;
      if (MODE == 0) Cf[idx] = v;
      else           Cb[idx] = (__bf16)v;
    }
  }
}

// ---------------------------------------------------------------- attention
// One workgroup (128 thr = 4 waves) per (batch, head). K staged by the async
// global->LDS engine (ASYNCcnt); V transposed into LDS; scores never leave LDS.
__global__ void attn_kernel(const __bf16* __restrict__ qb, const __bf16* __restrict__ kbuf,
                            const __bf16* __restrict__ vbuf, const int* __restrict__ amask,
                            __bf16* __restrict__ ctxb) {
  extern __shared__ char smem[];
  __bf16* Kl   = (__bf16*)smem;             // [256][64] bf16 : 32 KB
  __bf16* Vt   = (__bf16*)(smem + 32768);   // [64][256] bf16 : 32 KB (V transposed)
  float*  madd = (float*)(smem + 65536);    // [256] f32      : 1 KB
  const int wave = threadIdx.x >> 5;
  const int lane = threadIdx.x & 31;
  float*  sc = (float*)(smem + 66560) + wave * (16 * 256);               // 16 KB / wave
  __bf16* P  = (__bf16*)(smem + 66560 + 4 * 16384) + wave * (16 * 256);  // 8 KB / wave
  const int b = blockIdx.x / NHc;
  const int h = blockIdx.x % NHc;
  const int tid = threadIdx.x;

  // stage K via async tensor path: 2048 x b128 copies, memory -> LDS directly
  for (int idx = tid; idx < 256 * 8; idx += blockDim.x) {
    int tok = idx >> 3;
    int f8  = (idx & 7) * 8;
    size_t gbase = ((size_t)(b * Lc + tok)) * Hc + h * HDc + f8;
    async_load_b128((unsigned)(tok * HDc + f8) * 2u, kbuf + gbase);
    // V needs a transposed LDS layout -> classic load + scatter
    v8bf vv = *(const v8bf*)(vbuf + gbase);
#pragma unroll
    for (int j = 0; j < 8; ++j) Vt[(f8 + j) * 256 + tok] = vv[j];
  }
  for (int t = tid; t < 256; t += blockDim.x)
    madd[t] = (amask[b * Lc + t] > 0) ? 0.f : -10000.f;
  wait_asynccnt0();
  __syncthreads();

  const int r15 = lane & 15;
  const int hsel = (lane >> 4) & 1;

  for (int qt = wave; qt < 16; qt += 4) {  // 4 query tiles per wave, uniform loop
    const __bf16* qrow = qb + ((size_t)(b * Lc + qt * 16 + r15)) * Hc + h * HDc;
    v16bf qa0 = load_afrag(qrow, hsel * 16);
    v16bf qa1 = load_afrag(qrow, 32 + hsel * 16);
    // S = Q K^T / 8 + mask  -> LDS
    for (int nt = 0; nt < 16; ++nt) {
      const __bf16* krow = Kl + (nt * 16 + r15) * HDc;
      v16bf b0 = *(const v16bf*)(krow + hsel * 16);
      v16bf b1 = *(const v16bf*)(krow + 32 + hsel * 16);
      v8f acc = {};
      acc = wmma_bf16(qa0, b0, acc);
      acc = wmma_bf16(qa1, b1, acc);
      float ma = madd[nt * 16 + r15];
#pragma unroll
      for (int r = 0; r < 8; ++r)
        sc[(r + hsel * 8) * 256 + nt * 16 + r15] = acc[r] * 0.125f + ma;
    }
    __syncthreads();
    // row softmax, write bf16 probabilities
    if (lane < 16) {
      float* row = sc + lane * 256;
      float mx = -1e30f;
      for (int j = 0; j < 256; ++j) mx = fmaxf(mx, row[j]);
      float ssum = 0.f;
      for (int j = 0; j < 256; ++j) { float e = __expf(row[j] - mx); ssum += e; row[j] = e; }
      float inv = 1.f / ssum;
      __bf16* pr = P + lane * 256;
      for (int j = 0; j < 256; ++j) pr[j] = (__bf16)(row[j] * inv);
    }
    __syncthreads();
    // ctx = P V
    const __bf16* prow = P + r15 * 256;
    for (int nt = 0; nt < 4; ++nt) {
      const __bf16* vrow = Vt + (nt * 16 + r15) * 256;
      v8f acc = {};
#pragma unroll
      for (int ks = 0; ks < 8; ++ks) {
        v16bf af = load_afrag(prow, ks * 32 + hsel * 16);
        v16bf bf = *(const v16bf*)(vrow + ks * 32 + hsel * 16);
        acc = wmma_bf16(af, bf, acc);
      }
      int fcol = h * HDc + nt * 16 + r15;
#pragma unroll
      for (int r = 0; r < 8; ++r)
        ctxb[((size_t)(b * Lc + qt * 16 + r + hsel * 8)) * Hc + fcol] = (__bf16)acc[r];
    }
    __syncthreads();
  }
}

// ---------------------------------------------------------------- LayerNorm kernels

__device__ __forceinline__ float block_reduce_sum(float v, float* red) {
  int tid = threadIdx.x;
  red[tid] = v;
  __syncthreads();
  for (int s = 128; s > 0; s >>= 1) {
    if (tid < s) red[tid] += red[tid + s];
    __syncthreads();
  }
  float r = red[0];
  __syncthreads();
  return r;
}

__global__ void embed_ln_kernel(const int* __restrict__ ids, const int* __restrict__ tts,
                                const float* __restrict__ tok, const float* __restrict__ pos,
                                const float* __restrict__ typ, const float* __restrict__ g,
                                const float* __restrict__ bb, float* __restrict__ xo,
                                __bf16* __restrict__ xob) {
  __shared__ float red[256];
  int row = blockIdx.x;
  int p = row % Lc;
  int id = ids[row];
  int tt = tts[row];
  int tid = threadIdx.x;
  float v[3]; float s = 0.f;
#pragma unroll
  for (int j = 0; j < 3; ++j) {
    int c = tid + j * 256;
    v[j] = tok[(size_t)id * Hc + c] + pos[(size_t)p * Hc + c] + typ[(size_t)tt * Hc + c];
    s += v[j];
  }
  float mu = block_reduce_sum(s, red) * (1.f / Hc);
  float s2 = 0.f;
#pragma unroll
  for (int j = 0; j < 3; ++j) { float d = v[j] - mu; s2 += d * d; }
  float var = block_reduce_sum(s2, red) * (1.f / Hc);
  float rs = rsqrtf(var + 1e-12f);
#pragma unroll
  for (int j = 0; j < 3; ++j) {
    int c = tid + j * 256;
    float o = (v[j] - mu) * rs * g[c] + bb[c];
    xo[(size_t)row * Hc + c]  = o;
    xob[(size_t)row * Hc + c] = (__bf16)o;
  }
}

__global__ void residual_ln_kernel(const float* __restrict__ xin, const float* __restrict__ delta,
                                   const float* __restrict__ g, const float* __restrict__ bb,
                                   float* __restrict__ xo, __bf16* __restrict__ xob) {
  __shared__ float red[256];
  int row = blockIdx.x;
  int tid = threadIdx.x;
  float v[3]; float s = 0.f;
#pragma unroll
  for (int j = 0; j < 3; ++j) {
    int c = tid + j * 256;
    size_t idx = (size_t)row * Hc + c;
    v[j] = xin[idx] + delta[idx];
    s += v[j];
  }
  float mu = block_reduce_sum(s, red) * (1.f / Hc);
  float s2 = 0.f;
#pragma unroll
  for (int j = 0; j < 3; ++j) { float d = v[j] - mu; s2 += d * d; }
  float var = block_reduce_sum(s2, red) * (1.f / Hc);
  float rs = rsqrtf(var + 1e-12f);
#pragma unroll
  for (int j = 0; j < 3; ++j) {
    int c = tid + j * 256;
    float o = (v[j] - mu) * rs * g[c] + bb[c];
    size_t idx = (size_t)row * Hc + c;
    xo[idx]  = o;
    xob[idx] = (__bf16)o;
  }
}

// ---------------------------------------------------------------- compaction + CRF

__global__ void compact_kernel(const float* __restrict__ logits, const int* __restrict__ valid,
                               const int* __restrict__ amask, float* __restrict__ emc,
                               int* __restrict__ nmask) {
  __shared__ int tgt[Lc];
  int b = blockIdx.x;
  int tid = threadIdx.x;
  for (int t = tid; t < Lc; t += blockDim.x) nmask[b * Lc + t] = 0;
  for (int i = tid; i < Lc * TPc; i += blockDim.x) emc[(size_t)b * Lc * TPc + i] = 0.f;
  __syncthreads();
  if (tid == 0) {
    int pos = -1;
    for (int t = 0; t < Lc; ++t) {
      if (valid[b * Lc + t] == 1) { ++pos; tgt[t] = pos; }
      else tgt[t] = -1;
    }
  }
  __syncthreads();
  for (int t = tid; t < Lc; t += blockDim.x) {
    int d = tgt[t];
    if (d >= 0) {
      for (int j = 0; j < TPc; ++j)
        emc[((size_t)b * Lc + d) * TPc + j] = logits[((size_t)b * Lc + t) * TPc + j];
      nmask[b * Lc + d] = amask[b * Lc + t];
    }
  }
}

__global__ void zero_out_kernel(float* p) {
  if (threadIdx.x == 0 && blockIdx.x == 0) p[0] = 0.f;
}

__global__ void crf_kernel(const float* __restrict__ emc, const int* __restrict__ nmask,
                           const int* __restrict__ labels, const float* __restrict__ start,
                           const float* __restrict__ endv, const float* __restrict__ trans,
                           float* __restrict__ out) {
  __shared__ float alpha[Tc], nalpha[Tc];
  __shared__ int tags[Lc];
  __shared__ int msk[Lc];
  int b = blockIdx.x;
  int tid = threadIdx.x;
  for (int t = tid; t < Lc; t += blockDim.x) {
    int lb = labels[b * Lc + t];
    tags[t] = lb >= 0 ? lb : 0;
    msk[t] = nmask[b * Lc + t];
  }
  __syncthreads();
  const float* em = emc + (size_t)b * Lc * TPc;
  if (tid < Tc) alpha[tid] = start[tid] + em[tid];
  __syncthreads();
  for (int t = 1; t < Lc; ++t) {
    if (msk[t]) {  // uniform (shared) -> barriers safe
      if (tid < Tc) {
        float mx = -1e30f;
        for (int i = 0; i < Tc; ++i) mx = fmaxf(mx, alpha[i] + trans[i * Tc + tid]);
        float ssum = 0.f;
        for (int i = 0; i < Tc; ++i) ssum += __expf(alpha[i] + trans[i * Tc + tid] - mx);
        nalpha[tid] = mx + __logf(ssum) + em[t * TPc + tid];
      }
      __syncthreads();
      if (tid < Tc) alpha[tid] = nalpha[tid];
      __syncthreads();
    }
  }
  if (tid == 0) {
    float mx = -1e30f;
    for (int j = 0; j < Tc; ++j) mx = fmaxf(mx, alpha[j] + endv[j]);
    float ssum = 0.f;
    for (int j = 0; j < Tc; ++j) ssum += __expf(alpha[j] + endv[j] - mx);
    float logZ = mx + __logf(ssum);
    float sc = start[tags[0]] + em[tags[0]];
    int cnt = 0;
    for (int t = 0; t < Lc; ++t) cnt += msk[t] ? 1 : 0;
    for (int t = 1; t < Lc; ++t)
      if (msk[t]) sc += trans[tags[t - 1] * Tc + tags[t]] + em[t * TPc + tags[t]];
    sc += endv[tags[cnt - 1]];
    atomicAdd(out, logZ - sc);
  }
}

// ---------------------------------------------------------------- host orchestration

extern "C" void kernel_launch(void* const* d_in, const int* in_sizes, int n_in,
                              void* d_out, int out_size, void* d_ws, size_t ws_size,
                              hipStream_t stream) {
  (void)in_sizes; (void)n_in; (void)out_size; (void)ws_size;
  const int*   input_ids      = (const int*)d_in[0];
  const int*   attention_mask = (const int*)d_in[1];
  const int*   token_type_ids = (const int*)d_in[2];
  const int*   valid_mask     = (const int*)d_in[3];
  const int*   label_ids      = (const int*)d_in[4];
  const float* tok_emb  = (const float*)d_in[5];
  const float* pos_emb  = (const float*)d_in[6];
  const float* type_emb = (const float*)d_in[7];
  const float* emb_g    = (const float*)d_in[8];
  const float* emb_b    = (const float*)d_in[9];
  const float* Wq = (const float*)d_in[10]; const float* bq = (const float*)d_in[11];
  const float* Wk = (const float*)d_in[12]; const float* bk = (const float*)d_in[13];
  const float* Wv = (const float*)d_in[14]; const float* bv = (const float*)d_in[15];
  const float* Wo = (const float*)d_in[16]; const float* bo = (const float*)d_in[17];
  const float* ln1_g = (const float*)d_in[18]; const float* ln1_b = (const float*)d_in[19];
  const float* Wi = (const float*)d_in[20]; const float* bi = (const float*)d_in[21];
  const float* Wf = (const float*)d_in[22]; const float* bf_ = (const float*)d_in[23];
  const float* ln2_g = (const float*)d_in[24]; const float* ln2_b = (const float*)d_in[25];
  const float* cls_W = (const float*)d_in[26]; const float* cls_b = (const float*)d_in[27];
  const float* crf_start = (const float*)d_in[28];
  const float* crf_end   = (const float*)d_in[29];
  const float* crf_trans = (const float*)d_in[30];

  size_t cur = 0;
  char* wsb = (char*)d_ws;
  auto alloc = [&](size_t bytes) -> char* {
    char* p = wsb + cur;
    cur += (bytes + 255) & ~(size_t)255;
    return p;
  };
  const size_t HH = (size_t)Hc * Hc, HF = (size_t)Hc * Fc;
  __bf16* wqT  = (__bf16*)alloc(NLc * HH * 2);
  __bf16* wkT  = (__bf16*)alloc(NLc * HH * 2);
  __bf16* wvT  = (__bf16*)alloc(NLc * HH * 2);
  __bf16* woT  = (__bf16*)alloc(NLc * HH * 2);
  __bf16* wiT  = (__bf16*)alloc(NLc * HF * 2);
  __bf16* wfT  = (__bf16*)alloc(NLc * HF * 2);
  __bf16* clsT = (__bf16*)alloc((size_t)TPc * Hc * 2);
  float*  clsb = (float*)alloc(TPc * 4);
  float*  x_f32  = (float*)alloc((size_t)Mrows * Hc * 4);
  __bf16* x_bf   = (__bf16*)alloc((size_t)Mrows * Hc * 2);
  __bf16* q_bf   = (__bf16*)alloc((size_t)Mrows * Hc * 2);
  __bf16* k_bf   = (__bf16*)alloc((size_t)Mrows * Hc * 2);
  __bf16* v_bf   = (__bf16*)alloc((size_t)Mrows * Hc * 2);
  __bf16* ctx_bf = (__bf16*)alloc((size_t)Mrows * Hc * 2);
  float*  tmp    = (float*)alloc((size_t)Mrows * Hc * 4);
  __bf16* h_bf   = (__bf16*)alloc((size_t)Mrows * Fc * 2);
  float*  logits = (float*)alloc((size_t)Mrows * TPc * 4);
  float*  emc    = (float*)alloc((size_t)Mrows * TPc * 4);
  int*    nmask  = (int*)alloc((size_t)Bc * Lc * 4);

  // weight conversion (bf16 weight set ~57 MB -> resident in the 192 MB L2)
  for (int l = 0; l < NLc; ++l) {
    transpose_cast<<<Hc, 256, 0, stream>>>(Wq + l * HH, wqT + l * HH, Hc, Hc);
    transpose_cast<<<Hc, 256, 0, stream>>>(Wk + l * HH, wkT + l * HH, Hc, Hc);
    transpose_cast<<<Hc, 256, 0, stream>>>(Wv + l * HH, wvT + l * HH, Hc, Hc);
    transpose_cast<<<Hc, 256, 0, stream>>>(Wo + l * HH, woT + l * HH, Hc, Hc);
    transpose_cast<<<Fc, 256, 0, stream>>>(Wi + l * HF, wiT + l * HF, Hc, Fc);
    transpose_cast<<<Hc, 256, 0, stream>>>(Wf + l * HF, wfT + l * HF, Fc, Hc);
  }
  cls_prep<<<TPc, 256, 0, stream>>>(cls_W, cls_b, clsT, clsb);

  // embeddings + LN
  embed_ln_kernel<<<Mrows, 256, 0, stream>>>(input_ids, token_type_ids, tok_emb, pos_emb,
                                             type_emb, emb_g, emb_b, x_f32, x_bf);

  // 16x64 strip per wave (NT=4): groups = Mtiles * (Ntiles/4), 8 waves per block
  const int ghh = (Mrows / 16) * (Hc / 16 / 4) / 8;   // 384 blocks
  const int ghf = (Mrows / 16) * (Fc / 16 / 4) / 8;   // 1536 blocks
  const int gcl = (Mrows / 16) * 1 / 8;               // 32 blocks (NT=3, one group)
  const size_t attn_lds = 66560 + 4 * 16384 + 4 * 8192;  // 164,864 B of 320 KB/WGP

  for (int l = 0; l < NLc; ++l) {
    gemm_wmma<1, 4><<<ghh, 256, 0, stream>>>(x_bf, wqT + l * HH, bq + l * Hc, nullptr, q_bf,
                                             Hc, Hc / 16);
    gemm_wmma<1, 4><<<ghh, 256, 0, stream>>>(x_bf, wkT + l * HH, bk + l * Hc, nullptr, k_bf,
                                             Hc, Hc / 16);
    gemm_wmma<1, 4><<<ghh, 256, 0, stream>>>(x_bf, wvT + l * HH, bv + l * Hc, nullptr, v_bf,
                                             Hc, Hc / 16);
    attn_kernel<<<Bc * NHc, 128, attn_lds, stream>>>(q_bf, k_bf, v_bf, attention_mask,
                                                     ctx_bf);
    gemm_wmma<0, 4><<<ghh, 256, 0, stream>>>(ctx_bf, woT + l * HH, bo + l * Hc, tmp, nullptr,
                                             Hc, Hc / 16);
    residual_ln_kernel<<<Mrows, 256, 0, stream>>>(x_f32, tmp, ln1_g + l * Hc, ln1_b + l * Hc,
                                                  x_f32, x_bf);
    gemm_wmma<2, 4><<<ghf, 256, 0, stream>>>(x_bf, wiT + l * HF, bi + l * Fc, nullptr, h_bf,
                                             Hc, Fc / 16);
    gemm_wmma<0, 4><<<ghh, 256, 0, stream>>>(h_bf, wfT + l * HF, bf_ + l * Hc, tmp, nullptr,
                                             Fc, Hc / 16);
    residual_ln_kernel<<<Mrows, 256, 0, stream>>>(x_f32, tmp, ln2_g + l * Hc, ln2_b + l * Hc,
                                                  x_f32, x_bf);
  }

  // classifier (N padded 33 -> 48), compaction, CRF loss
  gemm_wmma<0, 3><<<gcl, 256, 0, stream>>>(x_bf, clsT, clsb, logits, nullptr, Hc, TPc / 16);
  compact_kernel<<<Bc, 256, 0, stream>>>(logits, valid_mask, attention_mask, emc, nmask);
  zero_out_kernel<<<1, 1, 0, stream>>>((float*)d_out);
  crf_kernel<<<Bc, 64, 0, stream>>>(emc, nmask, label_ids, crf_start, crf_end, crf_trans,
                                    (float*)d_out);
}
